// GPT_343597384140
// MI455X (gfx1250) — matrix operand order repaired and checked by reading, not compiled, not run on hardware
//
#include <hip/hip_runtime.h>
#include <hip/hip_bf16.h>
#include <math.h>

// GPT-2 small forward on gfx1250 (MI455X).
// - All GEMMs + attention matmuls via v_wmma_f32_16x16x32_bf16 (f32 accum).
// - bf16 tile copies (GEMM A-tiles, attention Q/K tiles) use the CDNA5 async
//   path: global_load_async_to_lds_b128 + s_wait_asynccnt (ASYNCcnt).
// - GEMM K-loop is double-buffered in LDS: prefetch tile t+1 while WMMAs
//   consume tile t.

typedef __bf16 bf16;
typedef __attribute__((ext_vector_type(8)))  __bf16 v8bf;
typedef __attribute__((ext_vector_type(16))) __bf16 v16bf;
typedef __attribute__((ext_vector_type(8)))  float  v8f;

#define CC   768
#define TT   1024
#define BB   4
#define MROW 4096   // B*T
#define NH   12
#define HD   64

__device__ __forceinline__ v8f wmma_bf16(v16bf a, v16bf b, v8f c) {
  // 8 args: (neg_a, A, neg_b, B, c_mod, C, reuse_a, reuse_b)
  return __builtin_amdgcn_wmma_f32_16x16x32_bf16(false, a, false, b, (short)0, c,
                                                 false, false);
}

// LDS byte offset of a __shared__ object: low 32 bits of the flat address
// (LDS aperture: addr = {SHARED_BASE, lds_offset[31:0]}).
__device__ __forceinline__ uint32_t lds_off(const void* p) {
  return (uint32_t)(uintptr_t)p;
}

// CDNA5 async copy: global -> LDS, 16 bytes per lane, tracked by ASYNCcnt.
__device__ __forceinline__ void async_copy_b128(uint32_t lds_addr, const void* gaddr) {
  asm volatile("global_load_async_to_lds_b128 %0, %1, off"
               :: "v"(lds_addr), "v"((uint64_t)(uintptr_t)gaddr)
               : "memory");
}
__device__ __forceinline__ void wait_async0() {
  asm volatile("s_wait_asynccnt 0x0" ::: "memory");
}

// A fragment (16x32 bf16, MxK): lane m = base+(l&15), g = l>>4.
// elems 0..7  -> K = g*8 + j        (contiguous 16B)
// elems 8..15 -> K = 16 + g*8 + j   (contiguous 16B)
__device__ __forceinline__ v16bf frag_a(const bf16* lds, int stride, int mBase, int k0) {
  int lane = threadIdx.x & 31;
  int m = mBase + (lane & 15);
  int g = lane >> 4;
  const bf16* p = lds + m * stride + k0;
  union { v16bf v; v8bf h[2]; } u;
  u.h[0] = *(const v8bf*)(p + g * 8);
  u.h[1] = *(const v8bf*)(p + 16 + g * 8);
  return u.v;
}

// B fragment (32x16 bf16, KxN) from K-transposed LDS tile ldsT[n][k]:
// lane n = base+(l&15), g = l>>4; elems 0..7 -> K = g*8 + j, elems 8..15 -> K = 16 + g*8 + j
__device__ __forceinline__ v16bf frag_b(const bf16* ldsT, int stride, int nBase, int k0) {
  int lane = threadIdx.x & 31;
  int n = nBase + (lane & 15);
  int g = lane >> 4;
  const bf16* p = ldsT + n * stride + k0;
  union { v16bf v; v8bf h[2]; } u;
  u.h[0] = *(const v8bf*)(p + g * 8);
  u.h[1] = *(const v8bf*)(p + 16 + g * 8);
  return u.v;
}

// ---------------------------------------------------------------- embedding
__global__ void __launch_bounds__(256)
embed_kernel(const int* __restrict__ idx, const float* __restrict__ wte,
             const float* __restrict__ wpe, float* __restrict__ x) {
  int bt  = blockIdx.x;          // 0..4095
  int t   = bt & (TT - 1);
  int tok = idx[bt];
  int tid = threadIdx.x;
#pragma unroll
  for (int i = 0; i < 3; ++i) {
    int c = tid + i * 256;
    x[(size_t)bt * CC + c] = wte[(size_t)tok * CC + c] + wpe[(size_t)t * CC + c];
  }
}

// ---------------------------------------------------------------- layernorm -> bf16
__global__ void __launch_bounds__(256)
ln_kernel(const float* __restrict__ x, const float* __restrict__ w,
          const float* __restrict__ b, bf16* __restrict__ out) {
  __shared__ float red[256];
  int row = blockIdx.x, tid = threadIdx.x;
  const float* xr = x + (size_t)row * CC;
  float v0 = xr[tid], v1 = xr[tid + 256], v2 = xr[tid + 512];
  red[tid] = v0 + v1 + v2;
  __syncthreads();
  for (int o = 128; o > 0; o >>= 1) { if (tid < o) red[tid] += red[tid + o]; __syncthreads(); }
  float mu = red[0] * (1.0f / CC);
  __syncthreads();
  float d0 = v0 - mu, d1 = v1 - mu, d2 = v2 - mu;
  red[tid] = d0 * d0 + d1 * d1 + d2 * d2;
  __syncthreads();
  for (int o = 128; o > 0; o >>= 1) { if (tid < o) red[tid] += red[tid + o]; __syncthreads(); }
  float rs = rsqrtf(red[0] * (1.0f / CC) + 1e-5f);
  bf16* orow = out + (size_t)row * CC;
  orow[tid]       = (bf16)(d0 * rs * w[tid]       + b[tid]);
  orow[tid + 256] = (bf16)(d1 * rs * w[tid + 256] + b[tid + 256]);
  orow[tid + 512] = (bf16)(d2 * rs * w[tid + 512] + b[tid + 512]);
}

// ---------------------------------------------------------------- WMMA GEMM
// out[M,N] = A[M,K](bf16) @ W[K,N](fp32, cvt->bf16 in tile loader) + bias
// 256 thr = 8 waves; block tile 128x64; wave -> 16x64 strip (4 accumulators).
// Double-buffered LDS; A-tile via async global->LDS copies.
template <bool GELU, bool RESIDUAL>
__global__ void __launch_bounds__(256)
gemm_kernel(const bf16* __restrict__ A, const float* __restrict__ W,
            const float* __restrict__ bias, float* __restrict__ outF,
            bf16* __restrict__ outB, int M, int N, int K) {
  __shared__ bf16 ldsA[2][128 * 32];
  __shared__ bf16 ldsBT[2][64 * 40];  // [n][k], stride 40 (pad, 16B aligned)
  const int tid  = threadIdx.x;
  const int lane = tid & 31;
  const int wave = tid >> 5;          // 0..7 -> M sub-tile
  const int m0 = blockIdx.y * 128;
  const int n0 = blockIdx.x * 64;

  // per-thread A-tile coordinates (two b128 chunks per thread)
  const int am0 = tid >> 2,           ak0 = (tid & 3) * 8;
  const int am1 = (tid + 256) >> 2,   ak1 = (tid & 3) * 8;   // (tid+256)&3 == tid&3

  v8f acc[4] = {};

  auto load_tile = [&](int buf, int k0) {
    // A tile 128x32 bf16: async global->LDS, 2 x b128 per thread
    async_copy_b128(lds_off(&ldsA[buf][am0 * 32 + ak0]),
                    A + (size_t)(m0 + am0) * K + k0 + ak0);
    async_copy_b128(lds_off(&ldsA[buf][am1 * 32 + ak1]),
                    A + (size_t)(m0 + am1) * K + k0 + ak1);
    // B tile 32x64 fp32 -> bf16, transposed into ldsBT[n][k]
#pragma unroll
    for (int i = 0; i < 2; ++i) {
      int q = tid + i * 256;            // 0..511 float4 chunks
      int k = q >> 4;                   // 0..31
      int n = (q & 15) * 4;             // 0..60
      const float4 wv = *(const float4*)(W + (size_t)(k0 + k) * N + n0 + n);
      ldsBT[buf][(n + 0) * 40 + k] = (bf16)wv.x;
      ldsBT[buf][(n + 1) * 40 + k] = (bf16)wv.y;
      ldsBT[buf][(n + 2) * 40 + k] = (bf16)wv.z;
      ldsBT[buf][(n + 3) * 40 + k] = (bf16)wv.w;
    }
  };

  load_tile(0, 0);
  wait_async0();
  __syncthreads();

  const int nk = K / 32;
  for (int t = 0; t < nk; ++t) {
    const int cur = t & 1;
    if (t + 1 < nk) load_tile(cur ^ 1, (t + 1) * 32);  // prefetch next tile
    v16bf af = frag_a(ldsA[cur], 32, wave * 16, 0);
#pragma unroll
    for (int j = 0; j < 4; ++j) {
      v16bf bfrag = frag_b(ldsBT[cur], 40, j * 16, 0);
      acc[j] = wmma_bf16(af, bfrag, acc[j]);
    }
    wait_async0();        // our async fills of the next buffer are in LDS
    __syncthreads();      // everyone done reading cur / writing next
  }

  // epilogue: C/D layout -> row = r + 8*(lane>>4), col = lane&15
  const int g = lane >> 4, nlo = lane & 15;
#pragma unroll
  for (int j = 0; j < 4; ++j) {
    int col = n0 + j * 16 + nlo;
    float bv = bias[col];
#pragma unroll
    for (int r = 0; r < 8; ++r) {
      int row = m0 + wave * 16 + r + 8 * g;
      float v = acc[j][r] + bv;
      if (GELU) {
        float c = v + 0.044715f * v * v * v;
        v = 0.5f * v * (1.0f + tanhf(0.7978845608028654f * c));
      }
      if (RESIDUAL) outF[(size_t)row * N + col] += v;
      else          outB[(size_t)row * N + col] = (bf16)v;
    }
  }
}

// ---------------------------------------------------------------- flash attention
// grid (B*H, T/64); 128 threads = 4 waves; wave owns 16 q-rows x 64 cols.
// Q and K tiles filled with async global->LDS copies; V transposed via VGPRs.
__global__ void __launch_bounds__(128)
attn_kernel(const bf16* __restrict__ qkv, bf16* __restrict__ y) {
  __shared__ bf16 ldsQ[64 * 64];
  __shared__ bf16 ldsK[64 * 64];
  __shared__ bf16 ldsVT[64 * 64];   // [d][kk]
  __shared__ bf16 ldsP[64 * 64];
  const int tid  = threadIdx.x;
  const int lane = tid & 31;
  const int wave = tid >> 5;        // 0..3
  const int bh = blockIdx.x;
  const int b  = bh / NH, h = bh % NH;
  const int qt = blockIdx.y;
  const size_t rs = 3 * CC;         // qkv row stride
  const bf16* qbase = qkv + (size_t)b * TT * rs + h * HD;
  const bf16* kbase = qbase + CC;
  const bf16* vbase = qbase + 2 * CC;

  // async-load Q tile 64x64 (waited before first use below)
#pragma unroll
  for (int i = 0; i < 4; ++i) {
    int e = tid + i * 128;          // 0..511 b128 chunks
    int q = e >> 3, d = (e & 7) * 8;
    async_copy_b128(lds_off(&ldsQ[q * 64 + d]),
                    qbase + (size_t)(qt * 64 + q) * rs + d);
  }

  const int g = lane >> 4, nlo = lane & 15;
  float mrow[8], lrow[8];
  v8f o[4] = {};
#pragma unroll
  for (int r = 0; r < 8; ++r) { mrow[r] = -1e30f; lrow[r] = 0.0f; }

  for (int kt = 0; kt <= qt; ++kt) {
#pragma unroll
    for (int i = 0; i < 4; ++i) {
      int e = tid + i * 128;
      int kk = e >> 3, d = (e & 7) * 8;
      async_copy_b128(lds_off(&ldsK[kk * 64 + d]),
                      kbase + (size_t)(kt * 64 + kk) * rs + d);
      v8bf vv = *(const v8bf*)(vbase + (size_t)(kt * 64 + kk) * rs + d);
#pragma unroll
      for (int u = 0; u < 8; ++u) ldsVT[(d + u) * 64 + kk] = vv[u];
    }
    wait_async0();
    __syncthreads();

    // S = Q @ K^T   (inner dim d=64 -> two 32-wide WMMA K-steps)
    v8f s[4] = {};
#pragma unroll
    for (int ks = 0; ks < 2; ++ks) {
      v16bf af = frag_a(ldsQ, 64, wave * 16, ks * 32);
#pragma unroll
      for (int j = 0; j < 4; ++j) {
        v16bf bfrag = frag_b(ldsK, 64, j * 16, ks * 32);  // ldsK[kk][d] == B^T layout
        s[j] = wmma_bf16(af, bfrag, s[j]);
      }
    }
    // scale + causal mask
#pragma unroll
    for (int j = 0; j < 4; ++j)
#pragma unroll
      for (int r = 0; r < 8; ++r) {
        float sv = s[j][r] * 0.125f;     // 1/sqrt(64)
        if (kt == qt) {
          int qloc = wave * 16 + r + 8 * g;
          int kloc = j * 16 + nlo;
          if (kloc > qloc) sv = -1e30f;
        }
        s[j][r] = sv;
      }
    // online softmax; rows live in 16-lane half-wave groups
#pragma unroll
    for (int r = 0; r < 8; ++r) {
      float tm = fmaxf(fmaxf(s[0][r], s[1][r]), fmaxf(s[2][r], s[3][r]));
#pragma unroll
      for (int off = 1; off < 16; off <<= 1) tm = fmaxf(tm, __shfl_xor(tm, off, 32));
      float mnew  = fmaxf(mrow[r], tm);
      float alpha = __expf(mrow[r] - mnew);
      mrow[r] = mnew;
      float ps = 0.0f;
#pragma unroll
      for (int j = 0; j < 4; ++j) { float p = __expf(s[j][r] - mnew); s[j][r] = p; ps += p; }
#pragma unroll
      for (int off = 1; off < 16; off <<= 1) ps += __shfl_xor(ps, off, 32);
      lrow[r] = lrow[r] * alpha + ps;
#pragma unroll
      for (int j = 0; j < 4; ++j) o[j][r] *= alpha;
    }
    // stage P through LDS (C-layout -> A-layout)
#pragma unroll
    for (int j = 0; j < 4; ++j)
#pragma unroll
      for (int r = 0; r < 8; ++r)
        ldsP[(wave * 16 + r + 8 * g) * 64 + j * 16 + nlo] = (bf16)s[j][r];
    __syncthreads();
    // O += P @ V
#pragma unroll
    for (int ks = 0; ks < 2; ++ks) {
      v16bf af = frag_a(ldsP, 64, wave * 16, ks * 32);
#pragma unroll
      for (int j = 0; j < 4; ++j) {
        v16bf bfrag = frag_b(ldsVT, 64, j * 16, ks * 32);
        o[j] = wmma_bf16(af, bfrag, o[j]);
      }
    }
    __syncthreads();
  }

  // normalize + store y[b, t, h*64+d] (bf16)
#pragma unroll
  for (int j = 0; j < 4; ++j)
#pragma unroll
    for (int r = 0; r < 8; ++r) {
      int qloc = wave * 16 + r + 8 * g;
      int d    = j * 16 + nlo;
      float v  = o[j][r] / lrow[r];
      y[(size_t)(b * TT + qt * 64 + qloc) * CC + h * HD + d] = (bf16)v;
    }
}

// ---------------------------------------------------------------- final LN + head matvec
__global__ void __launch_bounds__(256)
head_kernel(const float* __restrict__ x, const float* __restrict__ w,
            const float* __restrict__ b, const float* __restrict__ wh,
            float* __restrict__ out) {
  __shared__ float red[256];
  int row = blockIdx.x, tid = threadIdx.x;
  const float* xr = x + (size_t)row * CC;
  float v0 = xr[tid], v1 = xr[tid + 256], v2 = xr[tid + 512];
  red[tid] = v0 + v1 + v2;
  __syncthreads();
  for (int o = 128; o > 0; o >>= 1) { if (tid < o) red[tid] += red[tid + o]; __syncthreads(); }
  float mu = red[0] * (1.0f / CC);
  __syncthreads();
  float d0 = v0 - mu, d1 = v1 - mu, d2 = v2 - mu;
  red[tid] = d0 * d0 + d1 * d1 + d2 * d2;
  __syncthreads();
  for (int o = 128; o > 0; o >>= 1) { if (tid < o) red[tid] += red[tid + o]; __syncthreads(); }
  float rsv = rsqrtf(red[0] * (1.0f / CC) + 1e-5f);
  __syncthreads();
  float dot = (d0 * rsv * w[tid]       + b[tid])       * wh[tid]
            + (d1 * rsv * w[tid + 256] + b[tid + 256]) * wh[tid + 256]
            + (d2 * rsv * w[tid + 512] + b[tid + 512]) * wh[tid + 512];
  red[tid] = dot;
  __syncthreads();
  for (int o = 128; o > 0; o >>= 1) { if (tid < o) red[tid] += red[tid + o]; __syncthreads(); }
  if (tid == 0) out[row] = red[0];
}

// ---------------------------------------------------------------- driver
extern "C" void kernel_launch(void* const* d_in, const int* in_sizes, int n_in,
                              void* d_out, int out_size, void* d_ws, size_t ws_size,
                              hipStream_t stream) {
  (void)in_sizes; (void)n_in; (void)out_size; (void)ws_size;
  const int*   idx    = (const int*)  d_in[0];
  const float* wte    = (const float*)d_in[1];
  const float* wpe    = (const float*)d_in[2];
  const float* ln1_w  = (const float*)d_in[3];
  const float* ln1_b  = (const float*)d_in[4];
  const float* w_qkv  = (const float*)d_in[5];
  const float* b_qkv  = (const float*)d_in[6];
  const float* w_ao   = (const float*)d_in[7];
  const float* b_ao   = (const float*)d_in[8];
  const float* ln2_w  = (const float*)d_in[9];
  const float* ln2_b  = (const float*)d_in[10];
  const float* w_fc   = (const float*)d_in[11];
  const float* b_fc   = (const float*)d_in[12];
  const float* w_fp   = (const float*)d_in[13];
  const float* b_fp   = (const float*)d_in[14];
  const float* lnf_w  = (const float*)d_in[15];
  const float* lnf_b  = (const float*)d_in[16];
  const float* w_head = (const float*)d_in[17];
  float* out = (float*)d_out;

  // workspace layout (~70 MB)
  char* ws = (char*)d_ws;
  size_t off = 0;
  auto alloc = [&](size_t bytes) -> void* {
    void* p = ws + off;
    off = (off + bytes + 255) & ~(size_t)255;
    return p;
  };
  float* xf   = (float*)alloc((size_t)MROW * CC * 4);       // fp32 residual stream
  bf16* hbf   = (bf16*) alloc((size_t)MROW * CC * 2);       // LN output
  bf16* qkvbf = (bf16*) alloc((size_t)MROW * 3 * CC * 2);   // QKV
  bf16* ybf   = (bf16*) alloc((size_t)MROW * CC * 2);       // attn output
  bf16* hidbf = (bf16*) alloc((size_t)MROW * 4 * CC * 2);   // MLP hidden

  embed_kernel<<<MROW, 256, 0, stream>>>(idx, wte, wpe, xf);

  for (int l = 0; l < 12; ++l) {
    ln_kernel<<<MROW, 256, 0, stream>>>(xf, ln1_w + l * CC, ln1_b + l * CC, hbf);
    gemm_kernel<false, false><<<dim3(3 * CC / 64, MROW / 128), 256, 0, stream>>>(
        hbf, w_qkv + (size_t)l * CC * 3 * CC, b_qkv + (size_t)l * 3 * CC,
        nullptr, qkvbf, MROW, 3 * CC, CC);
    attn_kernel<<<dim3(BB * NH, TT / 64), 128, 0, stream>>>(qkvbf, ybf);
    gemm_kernel<false, true><<<dim3(CC / 64, MROW / 128), 256, 0, stream>>>(
        ybf, w_ao + (size_t)l * CC * CC, b_ao + (size_t)l * CC,
        xf, nullptr, MROW, CC, CC);
    ln_kernel<<<MROW, 256, 0, stream>>>(xf, ln2_w + l * CC, ln2_b + l * CC, hbf);
    gemm_kernel<true, false><<<dim3(4 * CC / 64, MROW / 128), 256, 0, stream>>>(
        hbf, w_fc + (size_t)l * CC * 4 * CC, b_fc + (size_t)l * 4 * CC,
        nullptr, hidbf, MROW, 4 * CC, CC);
    gemm_kernel<false, true><<<dim3(CC / 64, MROW / 128), 256, 0, stream>>>(
        hidbf, w_fp + (size_t)l * 4 * CC * CC, b_fp + (size_t)l * CC,
        xf, nullptr, MROW, CC, 4 * CC);
  }

  head_kernel<<<MROW, 256, 0, stream>>>(xf, lnf_w, lnf_b, w_head, out);
}
